// EdgeEncoder_19670950216141
// MI455X (gfx1250) — matrix operand order repaired and checked by reading, not compiled, not run on hardware
//
#include <hip/hip_runtime.h>
#include <math.h>

#define H    128
#define NP   64
#define NH   4

typedef __attribute__((ext_vector_type(16))) _Float16 v16h;
typedef __attribute__((ext_vector_type(8)))  float    v8f;

// ---------------- WMMA fragment helpers (wave32; this code's contiguous K order for A/B, not the ISA table order) ----------
// A 16x32 f16: lane<16 -> M=lane, VGPR v: K=2v,2v+1 ; lane>=16 -> M=lane-16, K=16+2v,..
//              (same K order as B).
__device__ inline v16h load_A16_lds(const _Float16* A, int row0, int k0, int ld) {
  int lane = threadIdx.x & 31;
  int r  = row0 + (lane & 15);
  int kh = ((lane >> 4) & 1) * 16;
  v16h out;
#pragma unroll
  for (int v = 0; v < 8; ++v) {
    int k = k0 + kh + 2*v;
    out[2*v]   = A[r*ld + k];
    out[2*v+1] = A[r*ld + k + 1];
  }
  return out;
}

__device__ inline v16h load_A16_f32src(const float* A, int row0, int k0, int ld) {
  int lane = threadIdx.x & 31;
  int r  = row0 + (lane & 15);
  int kh = ((lane >> 4) & 1) * 16;
  v16h out;
#pragma unroll
  for (int v = 0; v < 8; ++v) {
    int k = k0 + kh + 2*v;
    out[2*v]   = (_Float16)A[r*ld + k];
    out[2*v+1] = (_Float16)A[r*ld + k + 1];
  }
  return out;
}

// B 32x16 f16: lane<16 -> N=lane, VGPR v: K=2v,2v+1 ; lane>=16 -> N=lane-16, K=16+2v,..
__device__ inline v16h load_B16_lds(const _Float16* B, int k0, int n0, int ld) {
  int lane = threadIdx.x & 31;
  int n  = n0 + (lane & 15);
  int kh = ((lane >> 4) & 1) * 16;
  v16h out;
#pragma unroll
  for (int v = 0; v < 8; ++v) {
    int k = k0 + kh + 2*v;
    out[2*v]   = B[k*ld + n];
    out[2*v+1] = B[(k+1)*ld + n];
  }
  return out;
}

// B = (LDS matrix)^T : element(k,n) = M[n*ld + k]  (for q @ kbase^T)
__device__ inline v16h load_B16_ldsT(const _Float16* M, int k0, int n0, int ld) {
  int lane = threadIdx.x & 31;
  int n  = n0 + (lane & 15);
  int kh = ((lane >> 4) & 1) * 16;
  const _Float16* row = M + n*ld;
  v16h out;
#pragma unroll
  for (int v = 0; v < 8; ++v) {
    int k = k0 + kh + 2*v;
    out[2*v]   = row[k];
    out[2*v+1] = row[k+1];
  }
  return out;
}

// B = W^T from global f32 row-major W[n][k] (for x @ W^T GEMMs), convert to f16
__device__ inline v16h load_B16_gT(const float* __restrict__ W, int k0, int n0, int ld) {
  int lane = threadIdx.x & 31;
  int n  = n0 + (lane & 15);
  int kh = ((lane >> 4) & 1) * 16;
  const float* row = W + (long)n*ld;
  v16h out;
#pragma unroll
  for (int v = 0; v < 8; ++v) {
    int k = k0 + kh + 2*v;
    out[2*v]   = (_Float16)row[k];
    out[2*v+1] = (_Float16)row[k+1];
  }
  return out;
}

// C/D 16x16 f32: VGPR r: lane<16 -> (M=r, N=lane) ; lane>=16 -> (M=8+r, N=lane-16)
__device__ inline void store_C_f16(_Float16* C, v8f acc, int row0, int col0, int ld) {
  int lane = threadIdx.x & 31;
  int r0  = row0 + (((lane >> 4) & 1) << 3);
  int col = col0 + (lane & 15);
#pragma unroll
  for (int r = 0; r < 8; ++r) C[(r0+r)*ld + col] = (_Float16)acc[r];
}
__device__ inline void store_C_f32(float* C, v8f acc, int row0, int col0, int ld) {
  int lane = threadIdx.x & 31;
  int r0  = row0 + (((lane >> 4) & 1) << 3);
  int col = col0 + (lane & 15);
#pragma unroll
  for (int r = 0; r < 8; ++r) C[(r0+r)*ld + col] = acc[r];
}
__device__ inline v8f load_C_f32(const float* C, int row0, int col0, int ld) {
  int lane = threadIdx.x & 31;
  int r0  = row0 + (((lane >> 4) & 1) << 3);
  int col = col0 + (lane & 15);
  v8f acc;
#pragma unroll
  for (int r = 0; r < 8; ++r) acc[r] = C[(r0+r)*ld + col];
  return acc;
}

#define WMMA_F16(a, b, c) \
  __builtin_amdgcn_wmma_f32_16x16x32_f16(false, (a), false, (b), (short)0, (c), false, false)

// ---------------- Kernel 1: fused weights  Wk2 = Wk*W2, Wv2 = Wv*W2, folded biases ---
__global__ __launch_bounds__(256)
void prep_weights(const float* __restrict__ in_w, const float* __restrict__ in_b,
                  const float* __restrict__ W2,   const float* __restrict__ b2,
                  float* __restrict__ Wk2, float* __restrict__ Wv2,
                  float* __restrict__ ckb, float* __restrict__ cvb) {
  int t = blockIdx.x * 256 + threadIdx.x;
  if (t < 8192) {                       // Wk2[o][m] = sum_c Wk[o][c]*W2[c][m]
    int o = t >> 6, m = t & 63;
    const float* wr = in_w + (H + o) * H;
    float a = 0.f;
    for (int c = 0; c < H; ++c) a += wr[c] * W2[c*64 + m];
    Wk2[t] = a;
  } else if (t < 16384) {
    int u = t - 8192; int o = u >> 6, m = u & 63;
    const float* wr = in_w + (2*H + o) * H;
    float a = 0.f;
    for (int c = 0; c < H; ++c) a += wr[c] * W2[c*64 + m];
    Wv2[u] = a;
  } else if (t < 16512) {               // ckb = bk + Wk @ b2
    int o = t - 16384;
    const float* wr = in_w + (H + o) * H;
    float a = in_b[H + o];
    for (int c = 0; c < H; ++c) a += wr[c] * b2[c];
    ckb[o] = a;
  } else if (t < 16640) {               // cvb = bv + Wv @ b2
    int o = t - 16512;
    const float* wr = in_w + (2*H + o) * H;
    float a = in_b[2*H + o];
    for (int c = 0; c < H; ++c) a += wr[c] * b2[c];
    cvb[o] = a;
  }
}

// ---------------- LDS layout (time-multiplexed, 62.7KB) -----------------------------
#define OF_F16   0        // 16KB feats f16[64][128] -> scores f32[4][16][64] -> featsTile f16[16][128]
#define OF_KB    16384    // 16KB kbase f16[64][128] -> ctxAcc f32[16][128]
#define OF_VB    32768    // 16KB vbase f16[64][128] -> ctx16 f16[16][128]
#define OF_Q     49152    // 4KB  q f16[16][128]
#define OF_G     53248    // 8KB  g f16[16][4][64] -> wsum f16 -> attn_out f16[16][128]
#define OF_POS   61440    // 512B
#define OF_W1    61952    // 512B
#define OF_B1    62464    // 256B
#define SMEM_BYTES 62720

// ---------------- Kernel 2: one block = (scene, 16-query tile) -----------------------
__global__ __launch_bounds__(256, 1)
void edge_attn_main(const float* __restrict__ feats, const float* __restrict__ pos,
                    const float* __restrict__ W1,    const float* __restrict__ b1,
                    const float* __restrict__ in_w,  const float* __restrict__ in_b,
                    const float* __restrict__ out_w, const float* __restrict__ out_b,
                    const float* __restrict__ proj_w,const float* __restrict__ proj_b,
                    const float* __restrict__ Wk2,   const float* __restrict__ Wv2,
                    const float* __restrict__ ckb,   const float* __restrict__ cvb,
                    float* __restrict__ out) {
  __shared__ __align__(16) unsigned char smem[SMEM_BYTES];
  _Float16* sF     = (_Float16*)(smem + OF_F16);
  _Float16* sKb    = (_Float16*)(smem + OF_KB);
  _Float16* sVb    = (_Float16*)(smem + OF_VB);
  _Float16* sQ     = (_Float16*)(smem + OF_Q);
  _Float16* sG     = (_Float16*)(smem + OF_G);
  float*    sPos   = (float*)(smem + OF_POS);
  float*    sW1    = (float*)(smem + OF_W1);
  float*    sB1    = (float*)(smem + OF_B1);
  float*    sSc    = (float*)(smem + OF_F16);     // scores/attn f32 [4][16][64]
  float*    sCAcc  = (float*)(smem + OF_KB);      // ctx accumulator f32 [16][128]
  _Float16* sCtx16 = (_Float16*)(smem + OF_VB);   // ctx f16 [16][128]
  _Float16* sAout  = (_Float16*)(smem + OF_G);    // attn_out f16 [16][128]
  _Float16* sFt    = (_Float16*)(smem + OF_F16);  // feats tile f16 [16][128]

  const int t = threadIdx.x;
  const int wv = t >> 5;
  const int scene = blockIdx.x >> 2;
  const int qt = blockIdx.x & 3;
  const int R0 = scene * NP;       // first row of scene
  const int QR = R0 + qt * 16;     // first query row (global)

  // Phase 0: stage feats(f16), pos, W1/b1
  for (int c = t; c < NP*H; c += 256) sF[c] = (_Float16)feats[(long)R0*H + c];
  for (int c = t; c < NP*2; c += 256) sPos[c] = pos[(long)R0*2 + c];
  if (t < 128) sW1[t] = W1[t];
  if (t < 64)  sB1[t] = b1[t];
  __syncthreads();

  // Phase 1 (WMMA): kbase(64x128), vbase(64x128), q(16x128). 72 tiles over 8 waves.
  for (int tt = wv; tt < 72; tt += 8) {
    const float *Wp, *bvec; _Float16* dst; int arow, drow, ct;
    if (tt < 32)      { int u = tt;      arow = (u>>3)*16; drow = arow; ct = u&7; Wp = in_w + H*H;   bvec = ckb;  dst = sKb; }
    else if (tt < 64) { int u = tt - 32; arow = (u>>3)*16; drow = arow; ct = u&7; Wp = in_w + 2*H*H; bvec = cvb;  dst = sVb; }
    else              { int u = tt - 64; arow = qt*16;     drow = 0;    ct = u;   Wp = in_w;         bvec = in_b; dst = sQ;  }
    v8f acc = {};
#pragma unroll
    for (int ks = 0; ks < 4; ++ks) {
      v16h a = load_A16_lds(sF, arow, ks*32, H);
      v16h b = load_B16_gT(Wp, ks*32, ct*16, H);
      acc = WMMA_F16(a, b, acc);
    }
    float bv = bvec[ct*16 + (t & 15)];
#pragma unroll
    for (int r = 0; r < 8; ++r) acc[r] += bv;
    store_C_f16(dst, acc, drow, ct*16, H);
  }
  __syncthreads();   // sF dead; OF_F16 becomes scores

  // Phase 2 (VALU): g[i][h][m] = q[i,head_h] @ Wk2_head   (k-side fold)
  for (int c = t; c < 16*NH*64; c += 256) {
    int i = c >> 8, h = (c >> 6) & 3, m = c & 63;
    const float* wcol = Wk2 + (h*32)*64 + m;
    float a = 0.f;
#pragma unroll 8
    for (int o = 0; o < 32; ++o) a += (float)sQ[i*H + h*32 + o] * wcol[o*64];
    sG[c] = (_Float16)a;
  }
  // Phase 3 (WMMA): base scores per head = q_h @ kbase_h^T  (16 tiles)
  for (int tt = wv; tt < 16; tt += 8) {
    int h = tt >> 2, jt = tt & 3;
    v16h a = load_A16_lds(sQ, 0, h*32, H);
    v16h b = load_B16_ldsT(sKb, h*32, jt*16, H);
    v8f acc = {};
    acc = WMMA_F16(a, b, acc);
    store_C_f32(sSc + h*16*64, acc, 0, jt*16, 64);
  }
  __syncthreads();

  // Phase 4 (VALU): scores += hidden(i,j) . g[i,h] ; scale by 1/sqrt(dh)
  const float scale = 0.17677669529663687f;
  for (int p = t; p < 16*64; p += 256) {
    int i = p >> 6, j = p & 63;
    int ip = qt*16 + i;
    float rx = sPos[j*2]   - sPos[ip*2];
    float ry = sPos[j*2+1] - sPos[ip*2+1];
    const _Float16* gi = sG + i*256;
    float d0 = 0.f, d1 = 0.f, d2 = 0.f, d3 = 0.f;
#pragma unroll 4
    for (int m = 0; m < 64; ++m) {
      float hm = fmaxf(rx*sW1[m*2] + ry*sW1[m*2+1] + sB1[m], 0.f);
      d0 += hm * (float)gi[m];
      d1 += hm * (float)gi[64 + m];
      d2 += hm * (float)gi[128 + m];
      d3 += hm * (float)gi[192 + m];
    }
    sSc[(0*16+i)*64 + j] = (sSc[(0*16+i)*64 + j] + d0) * scale;
    sSc[(1*16+i)*64 + j] = (sSc[(1*16+i)*64 + j] + d1) * scale;
    sSc[(2*16+i)*64 + j] = (sSc[(2*16+i)*64 + j] + d2) * scale;
    sSc[(3*16+i)*64 + j] = (sSc[(3*16+i)*64 + j] + d3) * scale;
  }
  __syncthreads();

  // Phase 5 (VALU): softmax over j (64 rows = 4 heads x 16 queries)
  if (t < 64) {
    float* row = sSc + t*64;
    float mx = row[0];
    for (int j = 1; j < 64; ++j) mx = fmaxf(mx, row[j]);
    float sum = 0.f;
    for (int j = 0; j < 64; ++j) { float e = __expf(row[j] - mx); row[j] = e; sum += e; }
    float inv = 1.f / sum;
    for (int j = 0; j < 64; ++j) row[j] *= inv;
  }
  __syncthreads();

  // Phase 6 (VALU): wsum[i][h][m] = sum_j attn[h,i,j]*hidden(i,j,m)  (v-side fold)
  for (int c = t; c < 16*64; c += 256) {
    int i = c >> 6, m = c & 63;
    int ip = qt*16 + i;
    float ipx = sPos[ip*2], ipy = sPos[ip*2+1];
    float wa = sW1[m*2], wb = sW1[m*2+1], bb = sB1[m];
    float a0 = 0.f, a1 = 0.f, a2 = 0.f, a3 = 0.f;
    for (int j = 0; j < 64; ++j) {
      float hm = fmaxf((sPos[j*2]-ipx)*wa + (sPos[j*2+1]-ipy)*wb + bb, 0.f);
      a0 += sSc[(0*16+i)*64 + j] * hm;
      a1 += sSc[(1*16+i)*64 + j] * hm;
      a2 += sSc[(2*16+i)*64 + j] * hm;
      a3 += sSc[(3*16+i)*64 + j] * hm;
    }
    sG[i*256 +   0 + m] = (_Float16)a0;
    sG[i*256 +  64 + m] = (_Float16)a1;
    sG[i*256 + 128 + m] = (_Float16)a2;
    sG[i*256 + 192 + m] = (_Float16)a3;
  }
  __syncthreads();

  // Phase 7a (VALU): ctx_delta[i][o] = wsum[i][head(o)] . Wv2[o]
  for (int c = t; c < 16*H; c += 256) {
    int i = c >> 7, o = c & 127, h = o >> 5;
    const _Float16* wr = sG + i*256 + h*64;
    const float* v2 = Wv2 + o*64;
    float a = 0.f;
#pragma unroll 4
    for (int m = 0; m < 64; ++m) a += (float)wr[m] * v2[m];
    sCAcc[i*H + o] = a;
  }
  __syncthreads();

  // Phase 7b (WMMA): ctx += attn_h @ vbase_h  (8 tiles = 8 waves, C preloaded)
  {
    int h = wv >> 1, cn = wv & 1;
    int n0 = h*32 + cn*16;
    v8f acc = load_C_f32(sCAcc, 0, n0, H);
#pragma unroll
    for (int ks = 0; ks < 2; ++ks) {
      v16h a = load_A16_f32src(sSc + h*16*64, 0, ks*32, 64);
      v16h b = load_B16_lds(sVb, ks*32, n0, H);
      acc = WMMA_F16(a, b, acc);
    }
    store_C_f32(sCAcc, acc, 0, n0, H);
  }
  __syncthreads();

  // Phase 7c: ctx -> f16 (into old vbase slot); stage feats tile f16 (old scores slot)
  for (int c = t; c < 16*H; c += 256) sCtx16[c] = (_Float16)sCAcc[c];
  for (int c = t; c < 16*H; c += 256) sFt[c] = (_Float16)feats[(long)QR*H + c];
  __syncthreads();

  // Phase 8 (WMMA): attn_out = ctx @ out_w^T + out_b  -> f16 (old g slot)
  {
    int cn = wv;
    v8f acc = {};
#pragma unroll
    for (int ks = 0; ks < 4; ++ks) {
      v16h a = load_A16_lds(sCtx16, 0, ks*32, H);
      v16h b = load_B16_gT(out_w, ks*32, cn*16, H);
      acc = WMMA_F16(a, b, acc);
    }
    float bv = out_b[cn*16 + (t & 15)];
#pragma unroll
    for (int r = 0; r < 8; ++r) acc[r] += bv;
    store_C_f16(sAout, acc, 0, cn*16, H);
  }
  __syncthreads();

  // Phase 9 (WMMA): out = feats @ P1^T + attn_out @ P2^T + proj_b  (proj_w = [P1|P2])
  {
    int cn = wv;
    v8f acc = {};
#pragma unroll
    for (int ks = 0; ks < 4; ++ks) {
      v16h a = load_A16_lds(sFt, 0, ks*32, H);
      v16h b = load_B16_gT(proj_w, ks*32, cn*16, 2*H);
      acc = WMMA_F16(a, b, acc);
    }
#pragma unroll
    for (int ks = 0; ks < 4; ++ks) {
      v16h a = load_A16_lds(sAout, 0, ks*32, H);
      v16h b = load_B16_gT(proj_w + H, ks*32, cn*16, 2*H);
      acc = WMMA_F16(a, b, acc);
    }
    int lane = t & 31;
    float bv = proj_b[cn*16 + (lane & 15)];
    int r0 = ((lane >> 4) & 1) << 3;
    int col = cn*16 + (lane & 15);
#pragma unroll
    for (int r = 0; r < 8; ++r)
      out[(long)(QR + r0 + r)*H + col] = acc[r] + bv;
  }
}

// ---------------- host-side launch ---------------------------------------------------
extern "C" void kernel_launch(void* const* d_in, const int* in_sizes, int n_in,
                              void* d_out, int out_size, void* d_ws, size_t ws_size,
                              hipStream_t stream) {
  const float* feats  = (const float*)d_in[0];
  const float* pos    = (const float*)d_in[1];
  const float* W1     = (const float*)d_in[2];
  const float* b1     = (const float*)d_in[3];
  const float* W2     = (const float*)d_in[4];
  const float* b2     = (const float*)d_in[5];
  const float* in_w   = (const float*)d_in[6];
  const float* in_b   = (const float*)d_in[7];
  const float* out_w  = (const float*)d_in[8];
  const float* out_b  = (const float*)d_in[9];
  const float* proj_w = (const float*)d_in[10];
  const float* proj_b = (const float*)d_in[11];
  float* out = (float*)d_out;

  int total_rows = in_sizes[0] / H;     // S * N
  int S = total_rows / NP;

  float* ws    = (float*)d_ws;          // 66,560 bytes used
  float* wsWk2 = ws;
  float* wsWv2 = ws + 8192;
  float* wsCkb = ws + 16384;
  float* wsCvb = ws + 16512;

  prep_weights<<<65, 256, 0, stream>>>(in_w, in_b, W2, b2, wsWk2, wsWv2, wsCkb, wsCvb);
  edge_attn_main<<<S * 4, 256, 0, stream>>>(feats, pos, W1, b1, in_w, in_b,
                                            out_w, out_b, proj_w, proj_b,
                                            wsWk2, wsWv2, wsCkb, wsCvb, out);
}
